// DenseSNN_27650999452475
// MI455X (gfx1250) — matrix operand order repaired and checked by reading, not compiled, not run on hardware
//
#include <hip/hip_runtime.h>
#include <math.h>

typedef __attribute__((ext_vector_type(16))) __bf16 v16bf;
typedef __attribute__((ext_vector_type(8)))  float  v8f;

#define N_SZ   1024
#define IN_D   64
#define OUT_D  32
#define NSTEPS 33
#define ROWP   1048          // LDS row pad: 2096B = 524 dwords, bank-conflict-free for 16 rows
#define NTHREADS 512         // 16 waves
#define TPW    4             // 16-col n-tiles per wave (16 waves * 4 = 64 tiles = 1024 cols)
#define KTILES 32            // 1024 / 32

struct alignas(16) U128 { unsigned int x, y, z, w; };
union ABop { U128 q[2]; v16bf v; };

__device__ __forceinline__ unsigned short f2bf(float f) {
  union { float f; unsigned int u; } c; c.f = f;
  unsigned int u = c.u;
  return (unsigned short)((u + 0x7FFFu + ((u >> 16) & 1u)) >> 16);  // RNE
}
__device__ __forceinline__ float bf2f(unsigned short h) {
  union { unsigned int u; float f; } c; c.u = ((unsigned int)h) << 16;
  return c.f;
}

// Pre-pack R*kernel_h into hi/lo bf16 in the B-operand tile layout this kernel uses (k contiguous per lane half):
// tile (kt,nt) is 512 bf16: lane l holds K = kt*32 + (l/16)*16 + e, N = nt*16 + (l%16),
// at whi[(kt*64+nt)*512 + l*16 + e]  -> main loop reads 32 contiguous bytes per lane.
__global__ __launch_bounds__(256) void pack_weights(
    const float* __restrict__ kh, unsigned short* __restrict__ whi,
    unsigned short* __restrict__ wlo, float scale) {
  int idx = blockIdx.x * 256 + threadIdx.x;          // over 1024*1024 elements
  int k = idx >> 10, n = idx & (N_SZ - 1);
  float w = scale * kh[idx];
  unsigned short hi = f2bf(w);
  unsigned short lo = f2bf(w - bf2f(hi));
  int kt = k >> 5, kr = k & 31, nt = n >> 4;
  int lane = ((kr >> 4) << 4) | (n & 15);
  int e = kr & 15;
  int pos = ((kt * 64 + nt) << 9) + (lane << 4) + e;
  whi[pos] = hi;
  wlo[pos] = lo;
}

__global__ __launch_bounds__(NTHREADS) void snn_main(
    const float* __restrict__ x, const float* __restrict__ vm_in,
    const float* __restrict__ isyn_in, const float* __restrict__ rate_in,
    const unsigned char* __restrict__ spk_in, const float* __restrict__ kin,
    const unsigned short* __restrict__ whi, const unsigned short* __restrict__ wlo,
    const float* __restrict__ kout, float* __restrict__ out,
    float Rin, float aS, float aV, float aO, float Rout, float invdt) {
  __shared__ alignas(16) union {
    unsigned short spk[2][16][ROWP];   // bf16 bit patterns of spikes, A-operand friendly
    float rate[16][1032];              // reused after time loop for the output GEMM
  } sm;
  __shared__ float xs[16][IN_D];

  const int tid  = threadIdx.x;
  const int lane = tid & 31;
  const int wave = tid >> 5;
  const int l16  = lane & 15;
  const int hl   = lane >> 4;          // lane half: selects M/K sub-block per ISA layouts
  const int row0 = blockIdx.x * 16;    // 16 batch rows per workgroup

  // Register-resident state, C/D layout: lane(l16,hl), reg r -> m = r + hl*8, n = nt*16 + l16
  float vmv[TPW][8], isv[TPW][8], rtv[TPW][8], iin[TPW][8];
  v8f acc[TPW];

  // ---- load initial v_m / i_syn / rate ----
#pragma unroll
  for (int t = 0; t < TPW; t++) {
    int n = (wave * TPW + t) * 16 + l16;
#pragma unroll
    for (int r = 0; r < 8; r++) {
      int m = r + hl * 8;
      int off = (row0 + m) * N_SZ + n;
      vmv[t][r] = vm_in[off];
      isv[t][r] = isyn_in[off];
      rtv[t][r] = rate_in[off];
    }
  }

  // ---- initial spikes -> LDS buf0 (bf16 bit patterns), x -> LDS ----
  for (int i = tid; i < 16 * N_SZ; i += NTHREADS) {
    int m = i >> 10, c = i & (N_SZ - 1);
    sm.spk[0][m][c] = spk_in[(row0 + m) * N_SZ + c] ? (unsigned short)0x3F80 : (unsigned short)0;
  }
  for (int i = tid; i < 16 * IN_D; i += NTHREADS) {
    int m = i >> 6, j = i & 63;
    xs[m][j] = x[(row0 + m) * IN_D + j];
  }
  __syncthreads();

  // ---- i_in = R_in * (x @ kernel_in), once, plain FMA (tiny: 64-deep dots) ----
#pragma unroll
  for (int t = 0; t < TPW; t++) {
    int n = (wave * TPW + t) * 16 + l16;
    float a[8];
#pragma unroll
    for (int r = 0; r < 8; r++) a[r] = 0.f;
    for (int j = 0; j < IN_D; j++) {
      float kj = kin[j * N_SZ + n];
#pragma unroll
      for (int r = 0; r < 8; r++) a[r] += xs[r + hl * 8][j] * kj;
    }
#pragma unroll
    for (int r = 0; r < 8; r++) iin[t][r] = Rin * a[r];
  }

  // ---- time loop: i_spike = spike @ (R*kernel_h) via split-bf16 WMMA ----
  for (int s = 0; s < NSTEPS; s++) {
    const int cur = s & 1, nxt = cur ^ 1;
#pragma unroll
    for (int t = 0; t < TPW; t++)
#pragma unroll
      for (int r = 0; r < 8; r++) acc[t][r] = 0.f;

    for (int kt = 0; kt < KTILES; kt++) {
      // A operand (spike 16x32 bf16): lane l16 = row m; elems 0..7 at K=kb, 8..15 at K=kb+8
      ABop a;
      const int kb = kt * 32 + hl * 16;
      a.q[0] = *(const U128*)&sm.spk[cur][l16][kb];
      a.q[1] = *(const U128*)&sm.spk[cur][l16][kb + 8];
#pragma unroll
      for (int t = 0; t < TPW; t++) {
        int nt = wave * TPW + t;
        int base = ((kt * 64 + nt) << 9) + (lane << 4);
        ABop bh, bl;
        const U128* ph = (const U128*)(whi + base);
        bh.q[0] = ph[0]; bh.q[1] = ph[1];
        acc[t] = __builtin_amdgcn_wmma_f32_16x16x32_bf16(
            false, a.v, false, bh.v, (short)0, acc[t], false, false);
        const U128* pl = (const U128*)(wlo + base);
        bl.q[0] = pl[0]; bl.q[1] = pl[1];
        acc[t] = __builtin_amdgcn_wmma_f32_16x16x32_bf16(
            false, a.v, false, bl.v, (short)0, acc[t], false, false);
      }
    }

    // ---- elementwise state update + spike write to next LDS buffer ----
#pragma unroll
    for (int t = 0; t < TPW; t++) {
      int n = (wave * TPW + t) * 16 + l16;
#pragma unroll
      for (int r = 0; r < 8; r++) {
        float is  = isv[t][r] * aS + acc[t][r];
        float tgt = is + iin[t][r];
        float vm  = tgt + aV * (vmv[t][r] - tgt);
        bool  sp  = vm > 1.0f;
        vm = sp ? 0.f : vm;
        float spr = sp ? invdt : 0.f;
        float rt  = spr + aO * (rtv[t][r] - spr);
        isv[t][r] = is; vmv[t][r] = vm; rtv[t][r] = rt;
        sm.spk[nxt][r + hl * 8][n] = sp ? (unsigned short)0x3F80 : (unsigned short)0;
      }
    }
    __syncthreads();   // double-buffered: one barrier per step
  }

  // ---- outputs: (v_m, i_syn, rate, spike, out) concatenated as f32 ----
  float* o_vm = out;
  float* o_is = out + 4194304;
  float* o_rt = out + 8388608;
  float* o_sp = out + 12582912;
  float* o_ou = out + 16777216;
#pragma unroll
  for (int t = 0; t < TPW; t++) {
    int n = (wave * TPW + t) * 16 + l16;
#pragma unroll
    for (int r = 0; r < 8; r++) {
      int m = r + hl * 8;
      int off = (row0 + m) * N_SZ + n;
      o_vm[off] = vmv[t][r];
      o_is[off] = isv[t][r];
      o_rt[off] = rtv[t][r];
    }
  }
  const int fb = ((NSTEPS - 1) & 1) ^ 1;   // buffer holding final spikes
  for (int i = tid; i < 16 * N_SZ; i += NTHREADS) {
    int m = i >> 10, c = i & (N_SZ - 1);
    o_sp[(row0 + m) * N_SZ + c] = sm.spk[fb][m][c] ? 1.0f : 0.0f;
  }
  __syncthreads();                          // spike buffer consumed; safe to alias
#pragma unroll
  for (int t = 0; t < TPW; t++) {
    int n = (wave * TPW + t) * 16 + l16;
#pragma unroll
    for (int r = 0; r < 8; r++) sm.rate[r + hl * 8][n] = rtv[t][r];
  }
  __syncthreads();

  // out = R_out * (rate @ kernel_out): 16x32 per WG, rate broadcast from LDS
  {
    int m = tid >> 5, o = tid & 31;         // 512 threads = 16 rows x 32 cols
    float a = 0.f;
    for (int j = 0; j < N_SZ; j++) a += sm.rate[m][j] * kout[j * OUT_D + o];
    o_ou[(row0 + m) * OUT_D + o] = Rout * a;
  }
}

extern "C" void kernel_launch(void* const* d_in, const int* in_sizes, int n_in,
                              void* d_out, int out_size, void* d_ws, size_t ws_size,
                              hipStream_t stream) {
  const float*         x    = (const float*)d_in[0];
  const float*         vm   = (const float*)d_in[1];
  const float*         isyn = (const float*)d_in[2];
  const float*         rate = (const float*)d_in[3];
  const unsigned char* spk  = (const unsigned char*)d_in[4];
  const float*         kin  = (const float*)d_in[5];
  const float*         kh   = (const float*)d_in[6];
  const float*         kout = (const float*)d_in[7];
  float* out = (float*)d_out;

  unsigned short* whi = (unsigned short*)d_ws;                  // 2 MB
  unsigned short* wlo = whi + (size_t)N_SZ * N_SZ;              // 2 MB

  const float R_in  = 0.1f * 1.0f * 10.0f * sqrtf(2.0f / 64.0f);
  const float R     = 1.0f * 1.0f * (10.0f / 5.0f) * sqrtf(2.0f / 1024.0f);
  const float R_out = 5.0f * sqrtf(1.0f / 1024.0f);
  const float aS = expf(-0.5f / 5.0f);
  const float aV = expf(-0.5f / 10.0f);
  const float aO = expf(-0.5f / 10.0f);
  const float invdt = 1.0f / 0.5f;

  pack_weights<<<(N_SZ * N_SZ) / 256, 256, 0, stream>>>(kh, whi, wlo, R);
  snn_main<<<4096 / 16, NTHREADS, 0, stream>>>(x, vm, isyn, rate, spk, kin,
                                               whi, wlo, kout, out,
                                               R_in, aS, aV, aO, R_out, invdt);
}
